// lamformer_4002909520016
// MI455X (gfx1250) — matrix operand order repaired and checked by reading, not compiled, not run on hardware
//
#include <hip/hip_runtime.h>
#include <math.h>

#define Bn  16
#define Hn  56
#define Wn  56
#define Cc  192
#define NHn 6
#define HDn 32
#define Ln  (Hn*Wn)        // 3136
#define CHn (4*Cc)         // 768
#define Mn  (Bn*Ln)        // 50176

typedef __attribute__((ext_vector_type(16))) __bf16 v16bf;
typedef __attribute__((ext_vector_type(8)))  float  v8f;

static __device__ __forceinline__ float sigmoidf_(float x){ return 1.f/(1.f+__expf(-x)); }

// ---------------------------------------------------------------------------
// Elementwise helpers
// ---------------------------------------------------------------------------
__global__ void cast_bf16_kernel(const float* __restrict__ a, __bf16* __restrict__ o, size_t n){
  size_t i = (size_t)blockIdx.x*blockDim.x + threadIdx.x;
  if (i < n) o[i] = (__bf16)a[i];
}
__global__ void mul_cast_bf16_kernel(const float* __restrict__ a, const float* __restrict__ b,
                                     __bf16* __restrict__ o, size_t n){
  size_t i = (size_t)blockIdx.x*blockDim.x + threadIdx.x;
  if (i < n) o[i] = (__bf16)(a[i]*b[i]);
}
__global__ void gelu_cast_bf16_kernel(const float* __restrict__ a, __bf16* __restrict__ o, size_t n){
  size_t i = (size_t)blockIdx.x*blockDim.x + threadIdx.x;
  if (i < n){ float v = a[i]; o[i] = (__bf16)(0.5f*v*(1.f + erff(v*0.70710678118f))); }
}
__global__ void zero_f32_kernel(float* p, size_t n){
  size_t i = (size_t)blockIdx.x*blockDim.x + threadIdx.x;
  if (i < n) p[i] = 0.f;
}

// ---------------------------------------------------------------------------
// LayerNorm over last dim (Cn = 192 or 768). in = a (+ b optional).
// ---------------------------------------------------------------------------
__global__ __launch_bounds__(64) void ln_kernel(const float* a, const float* b,
                                                const float* __restrict__ g,
                                                const float* __restrict__ bb,
                                                float* outf, __bf16* outbf, int Cn)
{
  const int row = blockIdx.x;
  const int tid = threadIdx.x;
  const float* ap = a + (size_t)row * Cn;
  const float* bp = b ? b + (size_t)row * Cn : nullptr;
  float s = 0.f, s2 = 0.f;
  for (int c = tid; c < Cn; c += 64){
    float v = ap[c] + (bp ? bp[c] : 0.f);
    s += v; s2 += v*v;
  }
  __shared__ float rs[64], rq[64];
  rs[tid] = s; rq[tid] = s2; __syncthreads();
  for (int o = 32; o > 0; o >>= 1){
    if (tid < o){ rs[tid] += rs[tid+o]; rq[tid] += rq[tid+o]; }
    __syncthreads();
  }
  const float mean = rs[0] / (float)Cn;
  const float var  = rq[0] / (float)Cn - mean*mean;
  const float inv  = rsqrtf(var + 1e-5f);
  for (int c = tid; c < Cn; c += 64){
    float v = ap[c] + (bp ? bp[c] : 0.f);
    float y = (v - mean) * inv * g[c] + bb[c];
    if (outf)  outf[(size_t)row*Cn + c] = y;
    if (outbf) outbf[(size_t)row*Cn + c] = (__bf16)y;
  }
}

// ---------------------------------------------------------------------------
// Depthwise 3x3 'SAME' conv + bias (+ optional silu) (+ optional residual)
// ---------------------------------------------------------------------------
__global__ void dwconv_kernel(const float* __restrict__ xin, const float* __restrict__ wgt,
                              const float* __restrict__ bias, const float* res,
                              float* out, int Cn, int act)
{
  const int p = blockIdx.x;
  const int c = threadIdx.x;
  const int ww = p % Wn;
  const int hh = (p / Wn) % Hn;
  float acc = bias[c];
  for (int dy = -1; dy <= 1; ++dy){
    const int yy = hh + dy;
    if (yy < 0 || yy >= Hn) continue;
    for (int dx = -1; dx <= 1; ++dx){
      const int xx = ww + dx;
      if (xx < 0 || xx >= Wn) continue;
      acc += xin[(size_t)(p + dy*Wn + dx) * Cn + c] * wgt[((dy+1)*3 + (dx+1)) * Cn + c];
    }
  }
  float y = acc;
  if (act == 1) y = acc * sigmoidf_(acc);
  if (res) y += res[(size_t)p * Cn + c];
  out[(size_t)p * Cn + c] = y;
}

// ---------------------------------------------------------------------------
// RoPE tables + apply
// ---------------------------------------------------------------------------
__global__ void rope_tables_kernel(float* __restrict__ cosT, float* __restrict__ sinT){
  const int p = blockIdx.x;
  const int i = threadIdx.x;          // 0..95
  const int hh = p / Wn, ww = p % Wn;
  const float kmax = (float)(Cc/4);
  float ang;
  if (i < Cc/4) ang = (float)hh * powf(10000.f, -(float)i / kmax);
  else          ang = (float)ww * powf(10000.f, -(float)(i - Cc/4) / kmax);
  cosT[(size_t)p*(Cc/2) + i] = cosf(ang);
  sinT[(size_t)p*(Cc/2) + i] = sinf(ang);
}

__global__ void rope_apply_kernel(const float* __restrict__ src, int stride, int off,
                                  const float* __restrict__ cosT, const float* __restrict__ sinT,
                                  float* outf, __bf16* outbf)
{
  const int bl = blockIdx.x;
  const int i  = threadIdx.x;         // 0..95
  const int l  = bl % Ln;
  const float xr = src[(size_t)bl*stride + off + 2*i];
  const float xi = src[(size_t)bl*stride + off + 2*i + 1];
  const float cc = cosT[(size_t)l*(Cc/2) + i];
  const float ss = sinT[(size_t)l*(Cc/2) + i];
  const float o0 = cc*xr - ss*xi;
  const float o1 = ss*xr + cc*xi;
  if (outf){  outf[(size_t)bl*Cc + 2*i] = o0;          outf[(size_t)bl*Cc + 2*i+1] = o1; }
  if (outbf){ outbf[(size_t)bl*Cc + 2*i] = (__bf16)o0; outbf[(size_t)bl*Cc + 2*i+1] = (__bf16)o1; }
}

// ---------------------------------------------------------------------------
// kmean / zden
// ---------------------------------------------------------------------------
__global__ __launch_bounds__(256) void kmean_kernel(const float* __restrict__ qk, float* __restrict__ kmean){
  const int bh = blockIdx.x;
  const int b = bh / NHn, h = bh % NHn;
  const int t = threadIdx.x;
  const int d = t & 31, chunk = t >> 5;
  float s = 0.f;
  for (int l = chunk; l < Ln; l += 8)
    s += qk[((size_t)b*Ln + l)*(2*Cc) + Cc + h*HDn + d];
  __shared__ float red[256];
  red[t] = s; __syncthreads();
  if (chunk == 0){
    float tot = 0.f;
    for (int cc2 = 0; cc2 < 8; ++cc2) tot += red[cc2*32 + d];
    kmean[(size_t)bh*HDn + d] = tot / (float)Ln;
  }
}

__global__ void zden_kernel(const float* __restrict__ qk, const float* __restrict__ kmean,
                            float* __restrict__ zden){
  const size_t idx = (size_t)blockIdx.x*blockDim.x + threadIdx.x;
  if (idx >= (size_t)Bn*Ln*NHn) return;
  const int h = (int)(idx % NHn);
  const size_t bl = idx / NHn;
  const int b = (int)(bl / Ln);
  const float* q  = qk + bl*(2*Cc) + h*HDn;
  const float* km = kmean + ((size_t)b*NHn + h)*HDn;
  float s = 0.f;
  for (int d = 0; d < HDn; ++d) s += q[d]*km[d];
  zden[idx] = 1.f / (s + 1e-6f);
}

// ---------------------------------------------------------------------------
// kv[b,h] (32x32) = (1/L) * sum_l kr[b,l,h,:]^T v[b,l,h,:]   via WMMA bf16
// ---------------------------------------------------------------------------
#define KVSPLIT 8
__global__ __launch_bounds__(32) void kv_wmma_kernel(const __bf16* __restrict__ kr,
                                                     const __bf16* __restrict__ v,
                                                     float* __restrict__ kvm)
{
  const int bh = blockIdx.x;
  const int sp = blockIdx.y;
  const int b = bh / NHn, h = bh % NHn;
  const int lane = threadIdx.x;
  const int mr = lane & 15, hi = lane >> 4;
  v8f a00 = {}, a01 = {}, a10 = {}, a11 = {};
  const int iters = Ln / 32;
  for (int it = sp; it < iters; it += KVSPLIT){
    const int l0 = it * 32;
    v16bf fa0, fa1, fb0, fb1;
#pragma unroll
    for (int j = 0; j < 8; ++j){
      const int ka = ((j & 4) ? (16 + (j - 4)*2) : (j*2)) + hi*8;   // A layout (interleaved)
      const int kb = ((j & 4) ? (16 + (j - 4)*2) : (j*2)) + hi*8;   // B layout (interleaved, same as A)
      const size_t ra0 = ((size_t)b*Ln + l0 + ka)*Cc + h*HDn;
      const size_t ra1 = ra0 + Cc;
      const size_t rb0 = ((size_t)b*Ln + l0 + kb)*Cc + h*HDn;
      const size_t rb1 = rb0 + Cc;
      fa0[2*j] = kr[ra0 + mr];        fa0[2*j+1] = kr[ra1 + mr];
      fa1[2*j] = kr[ra0 + 16 + mr];   fa1[2*j+1] = kr[ra1 + 16 + mr];
      fb0[2*j] = v[rb0 + mr];         fb0[2*j+1] = v[rb1 + mr];
      fb1[2*j] = v[rb0 + 16 + mr];    fb1[2*j+1] = v[rb1 + 16 + mr];
    }
    a00 = __builtin_amdgcn_wmma_f32_16x16x32_bf16(false, fa0, false, fb0, (short)0, a00, false, false);
    a01 = __builtin_amdgcn_wmma_f32_16x16x32_bf16(false, fa0, false, fb1, (short)0, a01, false, false);
    a10 = __builtin_amdgcn_wmma_f32_16x16x32_bf16(false, fa1, false, fb0, (short)0, a10, false, false);
    a11 = __builtin_amdgcn_wmma_f32_16x16x32_bf16(false, fa1, false, fb1, (short)0, a11, false, false);
  }
  const float sc = 1.0f / (float)Ln;
  float* o = kvm + (size_t)bh * (HDn*HDn);
  const int col = mr, rb = hi*8;
#pragma unroll
  for (int r = 0; r < 8; ++r){
    atomicAdd(&o[(rb + r)*HDn + col],            a00[r]*sc);
    atomicAdd(&o[(rb + r)*HDn + col + 16],       a01[r]*sc);
    atomicAdd(&o[(16 + rb + r)*HDn + col],       a10[r]*sc);
    atomicAdd(&o[(16 + rb + r)*HDn + col + 16],  a11[r]*sc);
  }
}

// att[b,l,h,e] = zden[b,l,h] * sum_d qr[b,l,h,d]*kv[b,h,d,e]
__global__ void att_kernel(const float* __restrict__ qr, const float* __restrict__ kvm,
                           const float* __restrict__ zden, float* __restrict__ att){
  const int bl = blockIdx.x;
  const int t = threadIdx.x;          // 0..191
  const int h = t >> 5, e = t & 31;
  const int b = bl / Ln;
  const float* q = qr + (size_t)bl*Cc + h*HDn;
  const float* K = kvm + ((size_t)b*NHn + h)*(HDn*HDn);
  float s = 0.f;
  for (int d = 0; d < HDn; ++d) s += q[d]*K[d*HDn + e];
  att[(size_t)bl*Cc + t] = s * zden[(size_t)bl*NHn + h];
}

// ---------------------------------------------------------------------------
// GEMM: Y[M,N] = act( X[M,K] @ W[K,N] + bias ) (+ res). bf16 in, f32 acc.
// 128x64 block tile, 8 waves, 32x32 wave tile (4 WMMA / K-step).
// X streamed via GLOBAL_LOAD_ASYNC_TO_LDS_B128 with LDS double buffering;
// W staged transposed via VGPRs with register double buffer.
// act: 0=none, 1=silu, 2=elu+1
// ---------------------------------------------------------------------------
#define GTM 128
#define GTN 64
#define GTK 32

static __device__ __forceinline__ void gemm_emit(float v, size_t o, int act,
                                                 const float* res, float* Yf, __bf16* Ybf){
  if (act == 1) v *= sigmoidf_(v);
  else if (act == 2) v = (v > 0.f) ? v + 1.f : __expf(v);
  if (res) v += res[o];
  if (Yf)  Yf[o]  = v;
  if (Ybf) Ybf[o] = (__bf16)v;
}

__global__ __launch_bounds__(256) void gemm_bf16_kernel(
    const __bf16* __restrict__ X, const __bf16* __restrict__ Wm,
    const float* __restrict__ bias, const float* res,
    float* Yf, __bf16* Ybf, int M, int N, int K, int act)
{
  __shared__ __bf16 sX[2][GTM][GTK];       // 2 x 8 KB
  __shared__ __bf16 sWT[2][GTN][GTK + 4];  // 2 x 4.5 KB (transposed, padded)
  const int bm = blockIdx.x * GTM;
  const int bn = blockIdx.y * GTN;
  const int tid = threadIdx.x;
  const int wid = tid >> 5;
  const int lane = tid & 31;
  const int wm = (wid & 3) * 32;          // wave row block (4 waves over 128 rows)
  const int wn = (wid >> 2) * 32;         // wave col block (2 waves over 64 cols)
  const int mr = lane & 15;
  const int hi = lane >> 4;

  v8f acc00 = {}, acc01 = {}, acc10 = {}, acc11 = {};

  // X staging: each thread async-copies 32 bytes (two B128) of the 128x32 tile
  const int lxr = (tid * 16) >> 5;        // row 0..127
  const int lxc = (tid * 16) & 31;        // col 0 or 16
  // W staging: each thread loads 8 consecutive bf16 along N at one K
  const int lwk = (tid * 8) >> 6;
  const int lwn = (tid * 8) & 63;

  const int iters = K / GTK;

  // prologue: async X(0) into buffer 0; W(0) into registers
  {
    const __bf16* gx = X + (size_t)(bm + lxr)*K + lxc;
    unsigned l0 = (unsigned)(size_t)&sX[0][lxr][lxc];   // LDS byte offset (flat addr low 32b)
    asm volatile("global_load_async_to_lds_b128 %0, %1, off" :: "v"(l0),       "v"(gx)     : "memory");
    asm volatile("global_load_async_to_lds_b128 %0, %1, off" :: "v"(l0 + 16u), "v"(gx + 8) : "memory");
  }
  uint4 wv = *(const uint4*)(Wm + (size_t)lwk*N + bn + lwn);

  for (int i = 0; i < iters; ++i){
    const int cur = i & 1;
    // scatter W tile for this iteration (transposed into [n][k])
    {
      const __bf16* wp = (const __bf16*)&wv;
#pragma unroll
      for (int t = 0; t < 8; ++t) sWT[cur][lwn + t][lwk] = wp[t];
    }
    asm volatile("s_wait_asynccnt 0x0" ::: "memory");   // our async X(i) has landed
    __syncthreads();
    if (i + 1 < iters){
      const int k0 = (i + 1) * GTK;
      const __bf16* gx = X + (size_t)(bm + lxr)*K + k0 + lxc;
      unsigned l1 = (unsigned)(size_t)&sX[1 - cur][lxr][lxc];
      asm volatile("global_load_async_to_lds_b128 %0, %1, off" :: "v"(l1),       "v"(gx)     : "memory");
      asm volatile("global_load_async_to_lds_b128 %0, %1, off" :: "v"(l1 + 16u), "v"(gx + 8) : "memory");
      wv = *(const uint4*)(Wm + (size_t)(k0 + lwk)*N + bn + lwn);
    }
    v16bf a0, a1, b0, b1;
#pragma unroll
    for (int j = 0; j < 8; ++j){
      const int ka = ((j & 4) ? (16 + (j - 4)*2) : (j*2)) + hi*8;   // A frag layout
      const int kb = ((j & 4) ? (16 + (j - 4)*2) : (j*2)) + hi*8;   // B frag layout (same as A)
      a0[2*j]   = sX[cur][wm + mr][ka];
      a0[2*j+1] = sX[cur][wm + mr][ka + 1];
      a1[2*j]   = sX[cur][wm + 16 + mr][ka];
      a1[2*j+1] = sX[cur][wm + 16 + mr][ka + 1];
      b0[2*j]   = sWT[cur][wn + mr][kb];
      b0[2*j+1] = sWT[cur][wn + mr][kb + 1];
      b1[2*j]   = sWT[cur][wn + 16 + mr][kb];
      b1[2*j+1] = sWT[cur][wn + 16 + mr][kb + 1];
    }
    acc00 = __builtin_amdgcn_wmma_f32_16x16x32_bf16(false, a0, false, b0, (short)0, acc00, false, false);
    acc01 = __builtin_amdgcn_wmma_f32_16x16x32_bf16(false, a0, false, b1, (short)0, acc01, false, false);
    acc10 = __builtin_amdgcn_wmma_f32_16x16x32_bf16(false, a1, false, b0, (short)0, acc10, false, false);
    acc11 = __builtin_amdgcn_wmma_f32_16x16x32_bf16(false, a1, false, b1, (short)0, acc11, false, false);
    __syncthreads();
  }

  const int row0 = bm + wm + hi*8;
  const int c0 = bn + wn + mr;
  const float bv0 = bias[c0];
  const float bv1 = bias[c0 + 16];
#pragma unroll
  for (int r = 0; r < 8; ++r){
    const size_t o00 = (size_t)(row0 + r)*N + c0;
    gemm_emit(acc00[r] + bv0, o00,      act, res, Yf, Ybf);
    gemm_emit(acc01[r] + bv1, o00 + 16, act, res, Yf, Ybf);
    const size_t o10 = (size_t)(row0 + 16 + r)*N + c0;
    gemm_emit(acc10[r] + bv0, o10,      act, res, Yf, Ybf);
    gemm_emit(acc11[r] + bv1, o10 + 16, act, res, Yf, Ybf);
  }
}

// ---------------------------------------------------------------------------
extern "C" void kernel_launch(void* const* d_in, const int* in_sizes, int n_in,
                              void* d_out, int out_size, void* d_ws, size_t ws_size,
                              hipStream_t stream)
{
  (void)in_sizes; (void)n_in; (void)out_size; (void)ws_size;
  const float* x_in    = (const float*)d_in[0];
  const float* cpe1_w  = (const float*)d_in[1];
  const float* cpe1_b  = (const float*)d_in[2];
  const float* norm1_g = (const float*)d_in[3];
  const float* norm1_b = (const float*)d_in[4];
  const float* act_w   = (const float*)d_in[5];
  const float* act_b   = (const float*)d_in[6];
  const float* in_w    = (const float*)d_in[7];
  const float* in_b    = (const float*)d_in[8];
  const float* dwc_w   = (const float*)d_in[9];
  const float* dwc_b   = (const float*)d_in[10];
  const float* qk_w    = (const float*)d_in[11];
  const float* qk_b    = (const float*)d_in[12];
  const float* lepe_w  = (const float*)d_in[13];
  const float* lepe_b  = (const float*)d_in[14];
  const float* out_w   = (const float*)d_in[15];
  const float* out_b   = (const float*)d_in[16];
  const float* cpe2_w  = (const float*)d_in[17];
  const float* cpe2_b  = (const float*)d_in[18];
  const float* norm2_g = (const float*)d_in[19];
  const float* norm2_b = (const float*)d_in[20];
  const float* fc1_w   = (const float*)d_in[21];
  const float* fc1_b   = (const float*)d_in[22];
  const float* mdw_w   = (const float*)d_in[23];
  const float* mdw_b   = (const float*)d_in[24];
  const float* n1_g    = (const float*)d_in[25];
  const float* n1_b    = (const float*)d_in[26];
  const float* n2_g    = (const float*)d_in[27];
  const float* n2_b    = (const float*)d_in[28];
  const float* n3_g    = (const float*)d_in[29];
  const float* n3_b    = (const float*)d_in[30];
  const float* fc2_w   = (const float*)d_in[31];
  const float* fc2_b   = (const float*)d_in[32];
  float* out = (float*)d_out;

  const size_t SZC  = (size_t)Mn * Cc;
  const size_t SZCH = (size_t)Mn * CHn;

  char* ws = (char*)d_ws;
  size_t off = 0;
  auto alloc = [&](size_t bytes) -> void* {
    void* p = ws + off;
    off += (bytes + 255) & ~(size_t)255;
    return p;
  };
  float*  x1      = (float*) alloc(SZC * 4);       // shortcut; later x3
  float*  act_res = (float*) alloc(SZC * 4);
  float*  h0      = (float*) alloc(SZC * 4);       // in_proj out; later qr
  float*  h1      = (float*) alloc(SZC * 4);       // v
  float*  qkbuf   = (float*) alloc(SZC * 4 * 2);   // q|k; later att / x2
  __bf16* xn_bf   = (__bf16*)alloc(SZC * 2);       // later tmp_bf
  __bf16* h1_bf   = (__bf16*)alloc(SZC * 2);       // later xn2_bf
  __bf16* kr_bf   = (__bf16*)alloc(SZC * 2);
  float*  cosT    = (float*) alloc((size_t)Ln * (Cc/2) * 4);
  float*  sinT    = (float*) alloc((size_t)Ln * (Cc/2) * 4);
  float*  kmean   = (float*) alloc((size_t)Bn * NHn * HDn * 4);
  float*  zden    = (float*) alloc((size_t)Bn * Ln * NHn * 4);
  float*  kvbuf   = (float*) alloc((size_t)Bn * NHn * HDn * HDn * 4);
  float*  f1      = (float*) alloc(SZCH * 4);      // later g_bf overlay
  float*  dwb     = (float*) alloc(SZCH * 4);      // dw -> r1 -> r2 -> r3
  __bf16* actw_bf = (__bf16*)alloc((size_t)Cc*Cc*2);
  __bf16* inw_bf  = (__bf16*)alloc((size_t)Cc*Cc*2);
  __bf16* qkw_bf  = (__bf16*)alloc((size_t)Cc*2*Cc*2);
  __bf16* outw_bf = (__bf16*)alloc((size_t)Cc*Cc*2);
  __bf16* fc1w_bf = (__bf16*)alloc((size_t)Cc*CHn*2);
  __bf16* fc2w_bf = (__bf16*)alloc((size_t)CHn*Cc*2);

  float*  attb   = qkbuf;           // reuse (q/k dead by then)
  float*  x2     = qkbuf + SZC;     // reuse
  float*  x3     = x1;              // reuse (shortcut dead after out_proj)
  __bf16* tmp_bf = xn_bf;           // reuse
  __bf16* xn2_bf = h1_bf;           // reuse
  __bf16* g_bf   = (__bf16*)f1;     // reuse (f1 dead after r3)

  auto cgrid = [](size_t n){ return (unsigned)((n + 255) / 256); };

  // weight casts (tiny, L2-resident for the whole M sweep)
  cast_bf16_kernel<<<cgrid((size_t)Cc*Cc),  256, 0, stream>>>(act_w, actw_bf, (size_t)Cc*Cc);
  cast_bf16_kernel<<<cgrid((size_t)Cc*Cc),  256, 0, stream>>>(in_w,  inw_bf,  (size_t)Cc*Cc);
  cast_bf16_kernel<<<cgrid((size_t)Cc*2*Cc),256, 0, stream>>>(qk_w,  qkw_bf,  (size_t)Cc*2*Cc);
  cast_bf16_kernel<<<cgrid((size_t)Cc*Cc),  256, 0, stream>>>(out_w, outw_bf, (size_t)Cc*Cc);
  cast_bf16_kernel<<<cgrid((size_t)Cc*CHn), 256, 0, stream>>>(fc1_w, fc1w_bf, (size_t)Cc*CHn);
  cast_bf16_kernel<<<cgrid((size_t)CHn*Cc), 256, 0, stream>>>(fc2_w, fc2w_bf, (size_t)CHn*Cc);

  rope_tables_kernel<<<Ln, Cc/2, 0, stream>>>(cosT, sinT);

  // x1 = x + cpe1(x)
  dwconv_kernel<<<Bn*Hn*Wn, Cc, 0, stream>>>(x_in, cpe1_w, cpe1_b, x_in, x1, Cc, 0);
  // xn = LN(x1) -> bf16
  ln_kernel<<<Mn, 64, 0, stream>>>(x1, nullptr, norm1_g, norm1_b, nullptr, xn_bf, Cc);
  // act_res = silu(xn @ act_w + b)
  gemm_bf16_kernel<<<dim3(Mn/GTM, Cc/GTN), 256, 0, stream>>>(xn_bf, actw_bf, act_b, nullptr, act_res, nullptr, Mn, Cc, Cc, 1);
  // h0 = xn @ in_w + b
  gemm_bf16_kernel<<<dim3(Mn/GTM, Cc/GTN), 256, 0, stream>>>(xn_bf, inw_bf, in_b, nullptr, h0, nullptr, Mn, Cc, Cc, 0);
  // h1 = silu(dwc(h0))
  dwconv_kernel<<<Bn*Hn*Wn, Cc, 0, stream>>>(h0, dwc_w, dwc_b, nullptr, h1, Cc, 1);
  cast_bf16_kernel<<<cgrid(SZC), 256, 0, stream>>>(h1, h1_bf, SZC);
  // qk = elu(h1 @ qk_w + b) + 1
  gemm_bf16_kernel<<<dim3(Mn/GTM, (2*Cc)/GTN), 256, 0, stream>>>(h1_bf, qkw_bf, qk_b, nullptr, qkbuf, nullptr, Mn, 2*Cc, Cc, 2);
  // kmean, zden (use un-roped q,k)
  kmean_kernel<<<Bn*NHn, 256, 0, stream>>>(qkbuf, kmean);
  zden_kernel<<<cgrid((size_t)Bn*Ln*NHn), 256, 0, stream>>>(qkbuf, kmean, zden);
  // qr (f32, reuses h0), kr (bf16)
  rope_apply_kernel<<<Mn, Cc/2, 0, stream>>>(qkbuf, 2*Cc, 0,  cosT, sinT, h0, nullptr);
  rope_apply_kernel<<<Mn, Cc/2, 0, stream>>>(qkbuf, 2*Cc, Cc, cosT, sinT, nullptr, kr_bf);
  // kv = (1/L) kr^T v  (WMMA, split-L atomic reduce)
  zero_f32_kernel<<<cgrid((size_t)Bn*NHn*HDn*HDn), 256, 0, stream>>>(kvbuf, (size_t)Bn*NHn*HDn*HDn);
  kv_wmma_kernel<<<dim3(Bn*NHn, KVSPLIT), 32, 0, stream>>>(kr_bf, h1_bf, kvbuf);
  // att = (qr @ kv) * zden
  att_kernel<<<Mn, Cc, 0, stream>>>(h0, kvbuf, zden, attb);
  // att += lepe(v)
  dwconv_kernel<<<Bn*Hn*Wn, Cc, 0, stream>>>(h1, lepe_w, lepe_b, attb, attb, Cc, 0);
  // tmp = att * act_res  (bf16)
  mul_cast_bf16_kernel<<<cgrid(SZC), 256, 0, stream>>>(attb, act_res, tmp_bf, SZC);
  // x2 = tmp @ out_w + b + shortcut
  gemm_bf16_kernel<<<dim3(Mn/GTM, Cc/GTN), 256, 0, stream>>>(tmp_bf, outw_bf, out_b, x1, x2, nullptr, Mn, Cc, Cc, 0);
  // x3 = x2 + cpe2(x2)
  dwconv_kernel<<<Bn*Hn*Wn, Cc, 0, stream>>>(x2, cpe2_w, cpe2_b, x2, x3, Cc, 0);
  // xn2 = LN(x3) -> bf16
  ln_kernel<<<Mn, 64, 0, stream>>>(x3, nullptr, norm2_g, norm2_b, nullptr, xn2_bf, Cc);
  // f1 = xn2 @ fc1_w + b
  gemm_bf16_kernel<<<dim3(Mn/GTM, CHn/GTN), 256, 0, stream>>>(xn2_bf, fc1w_bf, fc1_b, nullptr, f1, nullptr, Mn, CHn, Cc, 0);
  // dw = mdw(f1)
  dwconv_kernel<<<Bn*Hn*Wn, CHn, 0, stream>>>(f1, mdw_w, mdw_b, nullptr, dwb, CHn, 0);
  // r1 = LN(dw + f1); r2 = LN(r1 + f1); r3 = LN(r2 + f1)
  ln_kernel<<<Mn, 64, 0, stream>>>(dwb, f1, n1_g, n1_b, dwb, nullptr, CHn);
  ln_kernel<<<Mn, 64, 0, stream>>>(dwb, f1, n2_g, n2_b, dwb, nullptr, CHn);
  ln_kernel<<<Mn, 64, 0, stream>>>(dwb, f1, n3_g, n3_b, dwb, nullptr, CHn);
  // g = gelu(r3) bf16
  gelu_cast_bf16_kernel<<<cgrid(SZCH), 256, 0, stream>>>(dwb, g_bf, SZCH);
  // out = g @ fc2_w + b + x3
  gemm_bf16_kernel<<<dim3(Mn/GTM, Cc/GTN), 256, 0, stream>>>(g_bf, fc2w_bf, fc2_b, x3, out, nullptr, Mn, Cc, CHn, 0);
}
